// CMoELayer_4011499454965
// MI455X (gfx1250) — matrix operand order repaired and checked by reading, not compiled, not run on hardware
//
#include <hip/hip_runtime.h>
#include <hip/hip_bf16.h>
#include <stdint.h>

// ---------------------------------------------------------------------------
// MoE layer for MI455X (gfx1250, wave32, WMMA).
// D_MODEL=1024, D_FF=4096, N_EXPERTS=8, TOP_K=2, T=4096 tokens.
// bf16 WMMA (v_wmma_f32_16x16x32_bf16) with f32 accumulate.
// Block tile 128x256, wave tile 64x64 (4x4 fragments, 16 WMMA : 8 b128 + 64 u16 ds_load).
// A tiles staged with global_load_async_to_lds_b128 (ASYNCcnt),
// B tiles converted f32->bf16 with v_perm_b32 packing.
// B fragments double-buffered so ds_load latency hides under WMMA quads.
// ---------------------------------------------------------------------------

#define D_MODEL 1024
#define D_FF    4096
#define N_EXP   8
#define TOKENS  4096
#define SLOTS   (TOKENS * 2)

typedef __attribute__((ext_vector_type(16))) __bf16 v16bf;
typedef __attribute__((ext_vector_type(8)))  float  v8f;
typedef unsigned short u16;

union Frag32B { v16bf v; uint4 q[2]; };

// f32 -> bf16 (truncate)
__device__ __forceinline__ u16 f2bf(float a) {
  return (u16)(__float_as_uint(a) >> 16);
}
// pack two f32 into two bf16 with one v_perm_b32
__device__ __forceinline__ unsigned pk2bf(float a, float b) {
  return __builtin_amdgcn_perm(__float_as_uint(b), __float_as_uint(a),
                               0x07060302u);
}

// ---------------------------------------------------------------------------
// Router: one wave (32 lanes) per group of 32 tokens; lanes stride D_MODEL.
// ---------------------------------------------------------------------------
__global__ __launch_bounds__(256) void k_router(
    const float* __restrict__ x, const float* __restrict__ Wg,
    int* __restrict__ ti, float* __restrict__ tw,
    int* __restrict__ pos, int* __restrict__ counts)
{
  __shared__ float wg[D_MODEL * N_EXP];   // 32 KB
  const int tid = threadIdx.x;
  for (int i = tid; i < D_MODEL * N_EXP; i += 256) wg[i] = Wg[i];
  __syncthreads();

  const int lane = tid & 31;
  const int wid  = tid >> 5;
  const int wave_global = blockIdx.x * 8 + wid;   // 16 blocks * 8 waves = 128

  for (int tt = 0; tt < 32; ++tt) {
    const int t = wave_global * 32 + tt;
    const float* xr = x + (size_t)t * D_MODEL;
    float a[N_EXP];
    #pragma unroll
    for (int e = 0; e < N_EXP; ++e) a[e] = 0.0f;
    for (int d = lane; d < D_MODEL; d += 32) {
      float xv = xr[d];
      #pragma unroll
      for (int e = 0; e < N_EXP; ++e) a[e] = fmaf(xv, wg[d * N_EXP + e], a[e]);
    }
    #pragma unroll
    for (int off = 16; off > 0; off >>= 1) {
      #pragma unroll
      for (int e = 0; e < N_EXP; ++e) a[e] += __shfl_xor(a[e], off, 32);
    }
    if (lane == 0) {
      int i0 = 0; float v0 = a[0];
      #pragma unroll
      for (int e = 1; e < N_EXP; ++e) if (a[e] > v0) { v0 = a[e]; i0 = e; }
      int i1 = -1; float v1 = -3.4e38f;
      #pragma unroll
      for (int e = 0; e < N_EXP; ++e)
        if (e != i0 && a[e] > v1) { v1 = a[e]; i1 = e; }
      float e1  = __expf(v1 - v0);
      float inv = 1.0f / (1.0f + e1);
      int p0 = atomicAdd(&counts[i0], 1);
      int p1 = atomicAdd(&counts[i1], 1);
      ti[2 * t]     = i0; ti[2 * t + 1] = i1;
      tw[2 * t]     = inv; tw[2 * t + 1] = e1 * inv;
      pos[2 * t]    = p0; pos[2 * t + 1] = p1;
    }
  }
}

__global__ void k_zero8(int* __restrict__ counts) {
  if (threadIdx.x < N_EXP) counts[threadIdx.x] = 0;
}

__global__ void k_scan(const int* __restrict__ counts, int* __restrict__ offs) {
  if (threadIdx.x == 0) {
    int s = 0;
    for (int e = 0; e < N_EXP; ++e) { offs[e] = s; s += counts[e]; }
    offs[N_EXP] = s;
  }
}

__global__ __launch_bounds__(256) void k_scatter(
    const int* __restrict__ ti, const float* __restrict__ tw,
    const int* __restrict__ pos, const int* __restrict__ offs,
    int* __restrict__ perm_tok, float* __restrict__ perm_wt,
    int* __restrict__ row_of)
{
  int j = blockIdx.x * 256 + threadIdx.x;      // [0, SLOTS)
  int e = ti[j];
  int row = offs[e] + pos[j];
  perm_tok[row] = j >> 1;
  perm_wt[row]  = tw[j];
  row_of[j]     = row;
}

// x (f32) -> x_bf16, 8 elements/thread
__global__ __launch_bounds__(256) void k_cvt_bf16(
    const float* __restrict__ src, u16* __restrict__ dst)
{
  int i = blockIdx.x * 256 + threadIdx.x;
  const float4* s = (const float4*)src + (size_t)i * 2;
  float4 f0 = s[0], f1 = s[1];
  uint4 o;
  o.x = pk2bf(f0.x, f0.y); o.y = pk2bf(f0.z, f0.w);
  o.z = pk2bf(f1.x, f1.y); o.w = pk2bf(f1.z, f1.w);
  ((uint4*)dst)[i] = o;
}

// ---------------------------------------------------------------------------
// WMMA mainloop: block tile 128(M) x 256(N), BK=32, 8 waves (2x4),
// wave tile 64x64 => 4x4 fragments of v_wmma_f32_16x16x32_bf16.
// A: bf16 rows (optionally gathered) -> LDS via global_load_async_to_lds_b128.
// B: f32 weights, converted to bf16 in registers, ds_store_b128.
// LDS: double-buffered, XOR-swizzled 16-byte granules. 48 KB total.
// ---------------------------------------------------------------------------
__device__ __forceinline__ void gemm_mainloop(
    uint4* __restrict__ As,   // 2 * 512  uint4 (128 x 32 bf16, dbl buffered)
    uint4* __restrict__ Bs,   // 2 * 1024 uint4 (32 x 256 bf16, dbl buffered)
    const u16* __restrict__ Ag, int lda,
    const int* __restrict__ gather, int rowBase, int mvalid,
    const float* __restrict__ Bg, int ldb, int nBase, int Ktot,
    v8f acc[4][4])
{
  const int tid  = threadIdx.x;
  const int lane = tid & 31;
  const int wid  = tid >> 5;
  const int wn   = wid & 3;    // 4 waves across N (64 cols each)
  const int wm   = wid >> 2;   // 2 waves across M (64 rows each)

  float4 rb[4][2];

  // Issue async DMA: global (bf16 tile rows) -> LDS, 16B per lane.
  auto issueA = [&](int k0, int buf) {
    uint4* A = As + buf * 512;
    #pragma unroll
    for (int i = 0; i < 2; ++i) {               // 512 granules / 256 thr
      int g = tid + i * 256;
      int r = g >> 2, gc = g & 3;
      uint4* dst = A + r * 4 + (gc ^ (r & 3));
      if (r < mvalid) {
        int arow = gather ? gather[rowBase + r] : (rowBase + r);
        const u16* p = Ag + (size_t)arow * lda + k0 + gc * 8;
        unsigned lds = (unsigned)(uintptr_t)dst;
        unsigned long long ga = (unsigned long long)(uintptr_t)p;
        asm volatile("global_load_async_to_lds_b128 %0, %1, off"
                     :: "v"(lds), "v"(ga) : "memory");
      } else {
        *dst = make_uint4(0u, 0u, 0u, 0u);      // zero-pad tail rows
      }
    }
  };
  auto fetchB = [&](int k0) {
    #pragma unroll
    for (int i = 0; i < 4; ++i) {               // 1024 granules / 256 thr
      int g = tid + i * 256;
      int k = g >> 5, gc = g & 31;
      const float* p = Bg + (size_t)(k0 + k) * ldb + nBase + gc * 8;
      rb[i][0] = ((const float4*)p)[0];
      rb[i][1] = ((const float4*)p)[1];
    }
  };
  auto commitB = [&](int buf) {
    uint4* B = Bs + buf * 1024;
    #pragma unroll
    for (int i = 0; i < 4; ++i) {
      int g = tid + i * 256;
      int k = g >> 5, gc = g & 31;
      uint4 v;
      v.x = pk2bf(rb[i][0].x, rb[i][0].y);
      v.y = pk2bf(rb[i][0].z, rb[i][0].w);
      v.z = pk2bf(rb[i][1].x, rb[i][1].y);
      v.w = pk2bf(rb[i][1].z, rb[i][1].w);
      B[k * 32 + (gc ^ (k & 31))] = v;
    }
  };

  issueA(0, 0);
  fetchB(0);
  commitB(0);

  const int KT = Ktot >> 5;
  const int l15 = lane & 15, lh = lane >> 4;
  for (int kt = 0; kt < KT; ++kt) {
    // Drain this wave's async DMA into buf (kt&1) before the barrier.
    asm volatile("s_wait_asynccnt 0x0" ::: "memory");
    __syncthreads();
    const bool more = (kt + 1) < KT;
    if (more) {
      issueA((kt + 1) << 5, (kt + 1) & 1);
      fetchB((kt + 1) << 5);
      if (kt + 2 < KT) {   // prefetch B strip two steps ahead into L2
        const float* pf = Bg + (size_t)((kt + 2) << 5) * ldb + nBase +
                          (tid & 31) * 8;
        __builtin_prefetch(pf, 0, 2);
      }
    }

    const uint4* A = As + (kt & 1) * 512;
    const uint4* B = Bs + (kt & 1) * 1024;

    // A frags: lane(0-15)=row M, K 0-7 & 16-23; lane(16-31): K 8-15 & 24-31
    Frag32B a[4];
    #pragma unroll
    for (int mf = 0; mf < 4; ++mf) {
      int row = wm * 64 + mf * 16 + l15;
      a[mf].q[0] = A[row * 4 + ( lh      ^ (row & 3))];
      a[mf].q[1] = A[row * 4 + ((lh + 2) ^ (row & 3))];
    }
    // B frags double-buffered: load nf+1 while WMMA quad consumes nf.
    // N = lane%16; K 0-7 & 16-23 (lanes 0-15), K 8-15 & 24-31 (lanes 16-31).
    Frag32B b[2];
    auto loadB = [&](Frag32B& f, int n) {
      const u16* Bh = (const u16*)B;
      u16 t[16];
      #pragma unroll
      for (int s = 0; s < 16; ++s) {
        int k = (s < 8) ? (8 * lh + s) : (16 + 8 * lh + (s - 8));
        t[s] = Bh[(k * 32 + ((n >> 3) ^ k)) * 8 + (n & 7)];
      }
      __builtin_memcpy(&f, t, 32);
    };
    loadB(b[0], wn * 64 + l15);
    #pragma unroll
    for (int nf = 0; nf < 4; ++nf) {
      if (nf < 3) {
        loadB(b[(nf + 1) & 1], wn * 64 + (nf + 1) * 16 + l15);
      }
      #pragma unroll
      for (int mf = 0; mf < 4; ++mf)
        acc[mf][nf] = __builtin_amdgcn_wmma_f32_16x16x32_bf16(
            false, a[mf].v, false, b[nf & 1].v, (short)0, acc[mf][nf],
            false, false);
    }

    if (more) commitB((kt + 1) & 1);
  }
}

#define GEMM_PROLOG()                                                   \
  __shared__ uint4 As[2 * 512];                                         \
  __shared__ uint4 Bs[2 * 1024];                                        \
  v8f acc[4][4];                                                        \
  _Pragma("unroll")                                                     \
  for (int mf = 0; mf < 4; ++mf)                                        \
    _Pragma("unroll")                                                   \
    for (int nf = 0; nf < 4; ++nf) acc[mf][nf] = {};                    \
  const int lane = threadIdx.x & 31;                                    \
  const int wid  = threadIdx.x >> 5;                                    \
  const int wn   = wid & 3, wm = wid >> 2;                              \
  const int l15  = lane & 15, lh = lane >> 4;

// Shared expert, layer 1: h = relu(x @ Ws1 + bs1) -> bf16
__global__ __launch_bounds__(256) void k_ffn1_shared(
    const u16* __restrict__ xb, const float* __restrict__ Ws1,
    const float* __restrict__ bs1, u16* __restrict__ hsh)
{
  GEMM_PROLOG();
  const int rowStart = blockIdx.y * 128;
  const int nBase    = blockIdx.x * 256;
  gemm_mainloop(As, Bs, xb, D_MODEL, nullptr, rowStart, 128,
                Ws1, D_FF, nBase, D_MODEL, acc);
  #pragma unroll
  for (int nf = 0; nf < 4; ++nf) {
    int c = nBase + wn * 64 + nf * 16 + l15;
    float bias = bs1[c];
    #pragma unroll
    for (int mf = 0; mf < 4; ++mf)
      #pragma unroll
      for (int i = 0; i < 8; ++i) {
        int r = rowStart + wm * 64 + mf * 16 + i + 8 * lh;
        float v = fmaxf(acc[mf][nf][i] + bias, 0.0f);
        hsh[(size_t)r * D_FF + c] = f2bf(v);
      }
  }
}

// Shared expert, layer 2: out = h @ Ws2 + bs2 (plain store, full coverage)
__global__ __launch_bounds__(256) void k_ffn2_shared(
    const u16* __restrict__ hsh, const float* __restrict__ Ws2,
    const float* __restrict__ bs2, float* __restrict__ out)
{
  GEMM_PROLOG();
  const int rowStart = blockIdx.y * 128;
  const int nBase    = blockIdx.x * 256;
  gemm_mainloop(As, Bs, hsh, D_FF, nullptr, rowStart, 128,
                Ws2, D_MODEL, nBase, D_FF, acc);
  #pragma unroll
  for (int nf = 0; nf < 4; ++nf) {
    int c = nBase + wn * 64 + nf * 16 + l15;
    float bias = bs2[c];
    #pragma unroll
    for (int mf = 0; mf < 4; ++mf)
      #pragma unroll
      for (int i = 0; i < 8; ++i) {
        int r = rowStart + wm * 64 + mf * 16 + i + 8 * lh;
        out[(size_t)r * D_MODEL + c] = acc[mf][nf][i] + bias;
      }
  }
}

// Routed experts, layer 1: gathered rows of x -> h_perm (bf16)
__global__ __launch_bounds__(256) void k_ffn1_exp(
    const u16* __restrict__ xb, const float* __restrict__ W1,
    const float* __restrict__ b1, const int* __restrict__ counts,
    const int* __restrict__ offs, const int* __restrict__ perm_tok,
    u16* __restrict__ hperm)
{
  const int e = blockIdx.z;
  const int cnt = counts[e];
  const int rowStart = blockIdx.y * 128;
  if (rowStart >= cnt) return;                 // uniform early-exit
  int mv = cnt - rowStart; if (mv > 128) mv = 128;
  const int base = offs[e] + rowStart;

  GEMM_PROLOG();
  const int nBase = blockIdx.x * 256;
  gemm_mainloop(As, Bs, xb, D_MODEL, perm_tok, base, mv,
                W1 + (size_t)e * D_MODEL * D_FF, D_FF, nBase, D_MODEL, acc);
  #pragma unroll
  for (int nf = 0; nf < 4; ++nf) {
    int c = nBase + wn * 64 + nf * 16 + l15;
    float bias = b1[e * D_FF + c];
    #pragma unroll
    for (int mf = 0; mf < 4; ++mf)
      #pragma unroll
      for (int i = 0; i < 8; ++i) {
        int rl = wm * 64 + mf * 16 + i + 8 * lh;
        if (rl < mv) {
          float v = fmaxf(acc[mf][nf][i] + bias, 0.0f);
          hperm[(size_t)(base + rl) * D_FF + c] = f2bf(v);
        }
      }
  }
}

// Routed experts, layer 2: y_perm[slot] = w * (h_perm @ W2[e] + b2[e])
__global__ __launch_bounds__(256) void k_ffn2_exp(
    const u16* __restrict__ hperm, const float* __restrict__ W2,
    const float* __restrict__ b2, const int* __restrict__ counts,
    const int* __restrict__ offs, const float* __restrict__ perm_wt,
    float* __restrict__ yperm)
{
  const int e = blockIdx.z;
  const int cnt = counts[e];
  const int rowStart = blockIdx.y * 128;
  if (rowStart >= cnt) return;
  int mv = cnt - rowStart; if (mv > 128) mv = 128;
  const int base = offs[e] + rowStart;

  GEMM_PROLOG();
  const int nBase = blockIdx.x * 256;
  gemm_mainloop(As, Bs, hperm, D_FF, nullptr, base, mv,
                W2 + (size_t)e * D_FF * D_MODEL, D_MODEL, nBase, D_FF, acc);
  #pragma unroll
  for (int mf = 0; mf < 4; ++mf)
    #pragma unroll
    for (int i = 0; i < 8; ++i) {
      int rl = wm * 64 + mf * 16 + i + 8 * lh;
      if (rl < mv) {
        float w = perm_wt[base + rl];
        #pragma unroll
        for (int nf = 0; nf < 4; ++nf) {
          int c = nBase + wn * 64 + nf * 16 + l15;
          yperm[(size_t)(base + rl) * D_MODEL + c] =
              w * (acc[mf][nf][i] + b2[e * D_MODEL + c]);
        }
      }
    }
}

// out[t] += y_perm[row0(t)] + y_perm[row1(t)]   (deterministic combine)
__global__ __launch_bounds__(256) void k_combine(
    float* __restrict__ out, const float* __restrict__ yperm,
    const int* __restrict__ row_of)
{
  int gid = blockIdx.x * 256 + threadIdx.x;  // TOKENS * 256 threads
  int t  = gid >> 8;
  int d4 = (gid & 255) * 4;
  int r0 = row_of[2 * t], r1 = row_of[2 * t + 1];
  float4 o = *(float4*)(out + (size_t)t * D_MODEL + d4);
  float4 a = *(const float4*)(yperm + (size_t)r0 * D_MODEL + d4);
  float4 b = *(const float4*)(yperm + (size_t)r1 * D_MODEL + d4);
  o.x += a.x + b.x; o.y += a.y + b.y; o.z += a.z + b.z; o.w += a.w + b.w;
  *(float4*)(out + (size_t)t * D_MODEL + d4) = o;
}

// ---------------------------------------------------------------------------
extern "C" void kernel_launch(void* const* d_in, const int* in_sizes, int n_in,
                              void* d_out, int out_size, void* d_ws, size_t ws_size,
                              hipStream_t stream)
{
  (void)in_sizes; (void)n_in; (void)out_size; (void)ws_size;
  const float* x   = (const float*)d_in[0];
  const float* Wg  = (const float*)d_in[1];
  const float* W1  = (const float*)d_in[2];
  const float* b1  = (const float*)d_in[3];
  const float* W2  = (const float*)d_in[4];
  const float* b2  = (const float*)d_in[5];
  const float* Ws1 = (const float*)d_in[6];
  const float* bs1 = (const float*)d_in[7];
  const float* Ws2 = (const float*)d_in[8];
  const float* bs2 = (const float*)d_in[9];
  float* out = (float*)d_out;

  char* w = (char*)d_ws;
  auto alloc = [&](size_t bytes) -> char* {
    char* p = w; w += (bytes + 255) & ~(size_t)255; return p;
  };
  int*   ti       = (int*)  alloc(SLOTS * sizeof(int));
  float* tw       = (float*)alloc(SLOTS * sizeof(float));
  int*   pos      = (int*)  alloc(SLOTS * sizeof(int));
  int*   counts   = (int*)  alloc(64);
  int*   offs     = (int*)  alloc(64);
  int*   perm_tok = (int*)  alloc(SLOTS * sizeof(int));
  float* perm_wt  = (float*)alloc(SLOTS * sizeof(float));
  int*   row_of   = (int*)  alloc(SLOTS * sizeof(int));
  u16*   xb       = (u16*)  alloc((size_t)TOKENS * D_MODEL * 2);
  u16*   hperm    = (u16*)  alloc((size_t)SLOTS * D_FF * 2);
  u16*   hsh      = (u16*)  alloc((size_t)TOKENS * D_FF * 2);
  float* yperm    = (float*)alloc((size_t)SLOTS * D_MODEL * 4);

  k_zero8  <<<1, 64, 0, stream>>>(counts);
  k_router <<<16, 256, 0, stream>>>(x, Wg, ti, tw, pos, counts);
  k_scan   <<<1, 32, 0, stream>>>(counts, offs);
  k_scatter<<<SLOTS / 256, 256, 0, stream>>>(ti, tw, pos, offs,
                                             perm_tok, perm_wt, row_of);
  k_cvt_bf16<<<(TOKENS * D_MODEL / 8) / 256, 256, 0, stream>>>(x, xb);

  // Shared expert
  k_ffn1_shared<<<dim3(D_FF / 256, TOKENS / 128), 256, 0, stream>>>(
      xb, Ws1, bs1, hsh);
  k_ffn2_shared<<<dim3(D_MODEL / 256, TOKENS / 128), 256, 0, stream>>>(
      hsh, Ws2, bs2, out);

  // Routed experts (fixed max grid; blocks early-exit on counts[e])
  k_ffn1_exp<<<dim3(D_FF / 256, TOKENS / 128, N_EXP), 256, 0, stream>>>(
      xb, W1, b1, counts, offs, perm_tok, hperm);
  k_ffn2_exp<<<dim3(D_MODEL / 256, TOKENS / 128, N_EXP), 256, 0, stream>>>(
      hperm, W2, b2, counts, offs, perm_wt, yperm);

  k_combine<<<TOKENS * 256 / 256, 256, 0, stream>>>(out, yperm, row_of);
}
